// MemoryNet_51196010169031
// MI455X (gfx1250) — compile-verified, matrix operand order checked by reading, not run on hardware
//
#include <hip/hip_runtime.h>
#include <cstdint>

#define NUM_MEM 4096
#define MEM_DIM 512
#define INP_DIM 1024
#define NROWS   32768

typedef __attribute__((ext_vector_type(16))) __bf16 v16bf;
typedef __attribute__((ext_vector_type(8)))  __bf16 v8bf;
typedef __attribute__((ext_vector_type(8)))  float  v8f;

// ---------------------------------------------------------------------------
// 16-bit A/B WMMA fragment = two contiguous 16-byte chunks per lane
// (K {0..7,16..23} for lanes 0-15, {8..15,24..31} for lanes 16-31; caller
// bakes the +8 hi-half element offset into the pointer).
// ---------------------------------------------------------------------------
__device__ __forceinline__ v16bf load_frag(const __bf16* p0) {
    v8bf lo = *(const v8bf*)(p0);
    v8bf hi = *(const v8bf*)(p0 + 16);
    v16bf f;
#pragma unroll
    for (int i = 0; i < 8; ++i) { f[i] = lo[i]; f[8 + i] = hi[i]; }
    return f;
}

__device__ __forceinline__ v8f wmma_bf16(v16bf a, v16bf b, v8f c) {
    return __builtin_amdgcn_wmma_f32_16x16x32_bf16(
        false, a, false, b, (short)0, c, false, false);
}

__device__ __forceinline__ float half16_sum(float v) {
#pragma unroll
    for (int i = 1; i < 16; i <<= 1) v += __shfl_xor(v, i, 32);
    return v;
}

// ---------------------------------------------------------------------------
// Stage 1: keyB = softmax(mem @ fk_w^T + fk_b)      [NUM_MEM, INP_DIM] bf16
//          valT = relu  (mem @ fv_w^T + fv_b)^T     [INP_DIM, NUM_MEM] bf16
// One block = 8 memory rows. Tiny stage (<2% of FLOPs) -> simple f32 GEMM.
// ---------------------------------------------------------------------------
__global__ __launch_bounds__(256) void kv_kernel(
    const float* __restrict__ mem,
    const float* __restrict__ fk_w, const float* __restrict__ fk_b,
    const float* __restrict__ fv_w, const float* __restrict__ fv_b,
    __bf16* __restrict__ keyB, __bf16* __restrict__ valT)
{
    __shared__ float smem[8][MEM_DIM];    // 16 KB
    __shared__ float slog[8][INP_DIM];    // 32 KB
    const int t    = threadIdx.x;
    const int lane = t & 31;
    const int wv   = t >> 5;              // 8 waves, wave w owns row w
    const int r0   = blockIdx.x * 8;

    for (int i = t; i < 8 * MEM_DIM; i += 256)
        smem[i >> 9][i & (MEM_DIM - 1)] = mem[(size_t)(r0 + (i >> 9)) * MEM_DIM + (i & (MEM_DIM - 1))];
    __syncthreads();

    // ---- key logits, softmax ----
#pragma unroll
    for (int cb = 0; cb < 4; ++cb) {
        const int c = cb * 256 + t;
        const float* w = fk_w + (size_t)c * MEM_DIM;
        float acc[8];
#pragma unroll
        for (int r = 0; r < 8; ++r) acc[r] = 0.f;
        for (int d = 0; d < MEM_DIM; ++d) {
            const float wd = w[d];
#pragma unroll
            for (int r = 0; r < 8; ++r) acc[r] = fmaf(smem[r][d], wd, acc[r]);
        }
        const float b = fk_b[c];
#pragma unroll
        for (int r = 0; r < 8; ++r) slog[r][c] = acc[r] + b;
    }
    __syncthreads();
    {
        float m = -3.0e38f;
        for (int j = lane; j < INP_DIM; j += 32) m = fmaxf(m, slog[wv][j]);
#pragma unroll
        for (int i = 1; i < 32; i <<= 1) m = fmaxf(m, __shfl_xor(m, i, 32));
        float s = 0.f;
        for (int j = lane; j < INP_DIM; j += 32) s += __expf(slog[wv][j] - m);
#pragma unroll
        for (int i = 1; i < 32; i <<= 1) s += __shfl_xor(s, i, 32);
        const float inv = 1.f / s;
        for (int j = lane; j < INP_DIM; j += 32)
            keyB[(size_t)(r0 + wv) * INP_DIM + j] = (__bf16)(__expf(slog[wv][j] - m) * inv);
    }
    __syncthreads();

    // ---- val logits, relu, transposed store ----
#pragma unroll
    for (int cb = 0; cb < 4; ++cb) {
        const int c = cb * 256 + t;
        const float* w = fv_w + (size_t)c * MEM_DIM;
        float acc[8];
#pragma unroll
        for (int r = 0; r < 8; ++r) acc[r] = 0.f;
        for (int d = 0; d < MEM_DIM; ++d) {
            const float wd = w[d];
#pragma unroll
            for (int r = 0; r < 8; ++r) acc[r] = fmaf(smem[r][d], wd, acc[r]);
        }
        const float b = fv_b[c];
#pragma unroll
        for (int r = 0; r < 8; ++r) slog[r][c] = acc[r] + b;
    }
    __syncthreads();
    for (int j = lane; j < INP_DIM; j += 32)
        valT[(size_t)j * NUM_MEM + (r0 + wv)] = (__bf16)fmaxf(slog[wv][j], 0.f);
}

// ---------------------------------------------------------------------------
// Fused attention: out = softmax(k @ keyB^T) @ val, single pass.
// keyB rows are probability vectors, so scores are convex combinations of
// N(0,1) entries of k: |s| <~ 6 -> exp() needs no max-subtraction.
// Accumulate unnormalized P @ valT plus per-row denominators; divide at end.
//
// One block = 32 k-rows x 1024 out-cols (two 16-row A tiles). Per 256-memory
// iteration each wave computes FOUR 16x16 S tiles (2 row-tiles x 2 col-tiles)
// so every A and B fragment feeds two WMMAs; then two P@valT phases where
// each valT fragment also feeds two WMMAs (both row tiles). Double-buffered
// P in LDS, one barrier per phase.
// ---------------------------------------------------------------------------
#define APITCH 1032   // 1024 + 8 bf16 pad -> conflict-free ds_load_b128
#define PPITCH 136    // 128 + 8 bf16 pad

__global__ __launch_bounds__(256) void attn_kernel(
    const float* __restrict__ kin, const __bf16* __restrict__ keyB,
    const __bf16* __restrict__ valT, float* __restrict__ out)
{
    __shared__ __bf16 sA[32][APITCH];        // ~66 KB, k tile in bf16
    __shared__ __bf16 sP[2][32][PPITCH];     // ~17 KB, double-buffered P tile
    __shared__ float  wd[8][32];             // per-wave denominator partials
    __shared__ float  sInv[32];
    const int t    = threadIdx.x;
    const int lane = t & 31;
    const int wv   = t >> 5;
    const int l15  = lane & 15;
    const int hi8  = (lane >> 4) * 8;        // frag K offset AND C-layout row offset
    const int r0   = blockIdx.x * 32;

    for (int i = t; i < 32 * INP_DIM; i += 256) {
        const int r = i >> 10, c = i & (INP_DIM - 1);
        sA[r][c] = (__bf16)kin[(size_t)(r0 + r) * INP_DIM + c];
    }
    __syncthreads();

    v8f acc0[8], acc1[8];                    // row-tile 0 / 1, 8 col-tiles each
#pragma unroll
    for (int j = 0; j < 8; ++j)
#pragma unroll
        for (int r = 0; r < 8; ++r) { acc0[j][r] = 0.f; acc1[j][r] = 0.f; }
    float den0[8], den1[8];
#pragma unroll
    for (int r = 0; r < 8; ++r) { den0[r] = 0.f; den1[r] = 0.f; }

    const __bf16* aptr0 = &sA[l15][hi8];
    const __bf16* aptr1 = &sA[16 + l15][hi8];
    for (int it = 0; it < NUM_MEM / 256; ++it) {
        const int base0 = it * 256;          // two 128-wide column chunks
        const __bf16* bptr0 = keyB + (size_t)(base0 + wv * 16 + l15) * INP_DIM + hi8;
        const __bf16* bptr1 = bptr0 + (size_t)128 * INP_DIM;
        __builtin_prefetch(bptr0 + (size_t)256 * INP_DIM, 0, 1);
        // ---- 4 S tiles: every fragment feeds two WMMAs ----
        v8f s00, s01, s10, s11;              // s[rowtile][coltile]
#pragma unroll
        for (int r = 0; r < 8; ++r) { s00[r] = 0.f; s01[r] = 0.f; s10[r] = 0.f; s11[r] = 0.f; }
#pragma unroll 4
        for (int kk = 0; kk < INP_DIM; kk += 32) {
            const v16bf a0 = load_frag(aptr0 + kk);
            const v16bf a1 = load_frag(aptr1 + kk);
            const v16bf b0 = load_frag(bptr0 + kk);
            const v16bf b1 = load_frag(bptr1 + kk);
            s00 = wmma_bf16(a0, b0, s00);
            s10 = wmma_bf16(a1, b0, s10);
            s01 = wmma_bf16(a0, b1, s01);
            s11 = wmma_bf16(a1, b1, s11);
        }
        // ---- two P@valT phases (double-buffered, 1 barrier per phase) ----
#pragma unroll
        for (int q = 0; q < 2; ++q) {
            const v8f& sa = q ? s01 : s00;
            const v8f& sb = q ? s11 : s10;
            const int membase = base0 + q * 128;
            const int buf = q;               // chunk parity == q
#pragma unroll
            for (int r = 0; r < 8; ++r) {
                const float p0 = __expf(sa[r]);
                const float p1 = __expf(sb[r]);
                den0[r] += p0;
                den1[r] += p1;
                sP[buf][r + hi8][wv * 16 + l15]      = (__bf16)p0;
                sP[buf][16 + r + hi8][wv * 16 + l15] = (__bf16)p1;
            }
            __syncthreads();
#pragma unroll
            for (int kk2 = 0; kk2 < 4; ++kk2) {
                const v16bf a0 = load_frag(&sP[buf][l15][kk2 * 32 + hi8]);
                const v16bf a1 = load_frag(&sP[buf][16 + l15][kk2 * 32 + hi8]);
#pragma unroll
                for (int j = 0; j < 8; ++j) {
                    const __bf16* vb = valT
                        + (size_t)(wv * 128 + j * 16 + l15) * NUM_MEM
                        + membase + kk2 * 32 + hi8;
                    const v16bf b = load_frag(vb);   // feeds both row tiles
                    acc0[j] = wmma_bf16(a0, b, acc0[j]);
                    acc1[j] = wmma_bf16(a1, b, acc1[j]);
                }
            }
            // no trailing barrier: next phase writes the OTHER buffer, and the
            // barrier above fences any previous readers of it
        }
    }

    // ---- merge denominators: 16-lane halves, then across 8 waves ----
#pragma unroll
    for (int r = 0; r < 8; ++r) { den0[r] = half16_sum(den0[r]); den1[r] = half16_sum(den1[r]); }
    if (lane == 0) {
#pragma unroll
        for (int r = 0; r < 8; ++r) { wd[wv][r]      = den0[r]; wd[wv][16 + r] = den1[r]; }
    }
    if (lane == 16) {
#pragma unroll
        for (int r = 0; r < 8; ++r) { wd[wv][8 + r]  = den0[r]; wd[wv][24 + r] = den1[r]; }
    }
    __syncthreads();
    if (t < 32) {
        float D = 0.f;
#pragma unroll
        for (int w = 0; w < 8; ++w) D += wd[w][t];
        sInv[t] = 1.f / D;
    }
    __syncthreads();

    float inv0[8], inv1[8];
#pragma unroll
    for (int r = 0; r < 8; ++r) { inv0[r] = sInv[r + hi8]; inv1[r] = sInv[16 + r + hi8]; }

    // ---- store out[32][1024] (C layout: VGPR r -> row r + hi8, col = l15) ----
#pragma unroll
    for (int j = 0; j < 8; ++j) {
        const int n0 = wv * 128 + j * 16;
#pragma unroll
        for (int r = 0; r < 8; ++r) {
            out[(size_t)(r0 + r + hi8) * INP_DIM + n0 + l15]      = acc0[j][r] * inv0[r];
            out[(size_t)(r0 + 16 + r + hi8) * INP_DIM + n0 + l15] = acc1[j][r] * inv1[r];
        }
    }
}

// ---------------------------------------------------------------------------
extern "C" void kernel_launch(void* const* d_in, const int* in_sizes, int n_in,
                              void* d_out, int out_size, void* d_ws, size_t ws_size,
                              hipStream_t stream) {
    const float* k    = (const float*)d_in[0];   // [N, INP_DIM]
    const float* mem  = (const float*)d_in[1];   // [NUM_MEM, MEM_DIM]
    const float* fk_w = (const float*)d_in[2];   // [INP_DIM, MEM_DIM]
    const float* fk_b = (const float*)d_in[3];   // [INP_DIM]
    const float* fv_w = (const float*)d_in[4];   // [INP_DIM, MEM_DIM]
    const float* fv_b = (const float*)d_in[5];   // [INP_DIM]
    float* out = (float*)d_out;                  // [N, INP_DIM] f32

    char* ws = (char*)d_ws;
    const size_t KEYB_BYTES = (size_t)NUM_MEM * INP_DIM * sizeof(__bf16);  // 8 MB
    __bf16* keyB = (__bf16*)(ws);
    __bf16* valT = (__bf16*)(ws + KEYB_BYTES);

    kv_kernel<<<NUM_MEM / 8, 256, 0, stream>>>(mem, fk_w, fk_b, fv_w, fv_b, keyB, valT);
    attn_kernel<<<NROWS / 32, 256, 0, stream>>>(k, keyB, valT, out);
}
